// WDL_38646115729863
// MI455X (gfx1250) — matrix operand order repaired and checked by reading, not compiled, not run on hardware
//
#include <hip/hip_runtime.h>
#include <hip/hip_bf16.h>
#include <math.h>

typedef __attribute__((ext_vector_type(16))) _Float16 v16h;
typedef __attribute__((ext_vector_type(8)))  _Float16 v8h;
typedef __attribute__((ext_vector_type(8)))  float    v8f;
typedef __attribute__((ext_vector_type(4)))  int      v4i;

static constexpr int BATCH = 16384;
static constexpr int SEQ   = 50;
static constexpr int DIM   = 16;
static constexpr int K0P   = 352;   // 326 padded up to 11*32 for WMMA K-steps

// ---------------------------------------------------------------------------
// CDNA5 async global->LDS staging (ASYNCcnt-tracked) with portable fallback.
// Builtin prototype (probe-learned): (v4i as(1)*, v4i as(3)*, Imm off, Imm cpol)
// ---------------------------------------------------------------------------
typedef __attribute__((address_space(1))) v4i g_v4i;
typedef __attribute__((address_space(3))) v4i s_v4i;

__device__ __forceinline__ void stage16(const _Float16* __restrict__ gsrc,
                                        _Float16* lds_dst) {
#if __has_builtin(__builtin_amdgcn_global_load_async_to_lds_b128)
    __builtin_amdgcn_global_load_async_to_lds_b128(
        (g_v4i*)gsrc, (s_v4i*)lds_dst, /*imm offset=*/0, /*cpol=*/0);
#else
    *(v8h*)lds_dst = *(const v8h*)gsrc;   // global_load_b128 + ds_store_b128
#endif
}

__device__ __forceinline__ void wait_async0() {
#if __has_builtin(__builtin_amdgcn_s_wait_asynccnt)
    __builtin_amdgcn_s_wait_asynccnt(0);
#else
    asm volatile("s_wait_asynccnt 0x0" ::: "memory");
#endif
}

// ---------------------------------------------------------------------------
// Stage 0: weight conversion f32 -> f16, with zero K-padding (layer 0 only)
// ---------------------------------------------------------------------------
__global__ void __launch_bounds__(256) wdl_w2h(const float* __restrict__ src,
                                               _Float16* __restrict__ dst,
                                               int N, int K, int Kp) {
    int i = blockIdx.x * blockDim.x + threadIdx.x;
    if (i >= N * Kp) return;
    int n = i / Kp, k = i - n * Kp;
    dst[i] = (k < K) ? (_Float16)src[n * K + k] : (_Float16)0.0f;
}

// ---------------------------------------------------------------------------
// Stage 1: embedding gather + history sum-pool + L2 normalize + wide logit.
// Block = 128 threads (4 waves). Each half-wave of 16 lanes owns one batch
// row; lane = embedding dim. Tables (~83MB total) are L2-resident.
// ---------------------------------------------------------------------------
__device__ __forceinline__ float emb_look(const float* __restrict__ t, int idx, int d) {
    return (idx == 0) ? 0.0f : t[idx * DIM + d];   // padding_idx = 0
}
__device__ __forceinline__ float hist_sum(const float* __restrict__ t,
                                          const int* __restrict__ idxs, int d) {
    float s = 0.0f;
    #pragma unroll 5
    for (int i = 0; i < SEQ; ++i) {
        int ix = idxs[i];
        if (ix) s += t[ix * DIM + d];
    }
    return s;
}

__global__ void __launch_bounds__(128) wdl_gather_normalize(
    const int* __restrict__ user_id, const int* __restrict__ district_id,
    const int* __restrict__ times_,  const int* __restrict__ shop_id,
    const int* __restrict__ item_id, const int* __restrict__ cat1,
    const int* __restrict__ msf,     const int* __restrict__ brand,
    const int* __restrict__ aoi,     const int* __restrict__ geo6,
    const int* __restrict__ shop_l,  const int* __restrict__ item_l,
    const int* __restrict__ cat1_l,  const int* __restrict__ msf_l,
    const int* __restrict__ brand_l, const int* __restrict__ aoi_l,
    const int* __restrict__ geo6_l,  const int* __restrict__ td_l,
    const int* __restrict__ ttype,   const int* __restrict__ ttype_l,
    const float* __restrict__ rank7, const float* __restrict__ rank30,
    const float* __restrict__ rank90,const float* __restrict__ hours,
    const float* __restrict__ price_l, const float* __restrict__ hours_l,
    const float* __restrict__ t_shop, const float* __restrict__ t_item,
    const float* __restrict__ t_cat1, const float* __restrict__ t_msf,
    const float* __restrict__ t_brand,const float* __restrict__ t_aoi,
    const float* __restrict__ t_geo6, const float* __restrict__ t_user,
    const float* __restrict__ t_district, const float* __restrict__ t_times,
    const float* __restrict__ t_td,   const float* __restrict__ t_tt,
    const float* __restrict__ t_ttl,
    const float* __restrict__ l_user, const float* __restrict__ l_shop,
    const float* __restrict__ l_item, const float* __restrict__ l_cat1,
    const float* __restrict__ l_shopl,const float* __restrict__ l_tt,
    _Float16* __restrict__ xf16, float* __restrict__ linlogit)
{
    const int tid  = threadIdx.x;
    const int lane = tid & 31;
    const int wav  = tid >> 5;
    const int hlf  = lane >> 4;
    const int d    = lane & 15;
    const int row  = blockIdx.x * 8 + wav * 2 + hlf;

    float v[20];
    v[0]  = emb_look(t_shop,  shop_id[row], d);
    v[1]  = hist_sum(t_shop,  shop_l  + row * SEQ, d);
    v[2]  = emb_look(t_item,  item_id[row], d);
    v[3]  = hist_sum(t_item,  item_l  + row * SEQ, d);
    v[4]  = emb_look(t_cat1,  cat1[row], d);
    v[5]  = hist_sum(t_cat1,  cat1_l  + row * SEQ, d);
    v[6]  = emb_look(t_msf,   msf[row], d);
    v[7]  = hist_sum(t_msf,   msf_l   + row * SEQ, d);
    v[8]  = emb_look(t_brand, brand[row], d);
    v[9]  = hist_sum(t_brand, brand_l + row * SEQ, d);
    v[10] = emb_look(t_aoi,   aoi[row], d);
    v[11] = hist_sum(t_aoi,   aoi_l   + row * SEQ, d);
    v[12] = emb_look(t_geo6,  geo6[row], d);
    v[13] = hist_sum(t_geo6,  geo6_l  + row * SEQ, d);
    v[14] = emb_look(t_user,  user_id[row], d);
    v[15] = emb_look(t_district, district_id[row], d);
    v[16] = emb_look(t_times, times_[row], d);
    v[17] = hist_sum(t_td,    td_l    + row * SEQ, d);
    v[18] = emb_look(t_tt,    ttype[row], d);
    v[19] = hist_sum(t_ttl,   ttype_l + row * SEQ, d);

    float sc[6] = { rank7[row], rank30[row], rank90[row],
                    hours[row], price_l[row], hours_l[row] };

    // squared norm over 326 features: 16-lane tree reduce + scalar terms
    float ss = 0.0f;
    #pragma unroll
    for (int i = 0; i < 20; ++i) ss += v[i] * v[i];
    #pragma unroll
    for (int m = 8; m >= 1; m >>= 1) ss += __shfl_xor(ss, m, 16);
    #pragma unroll
    for (int j = 0; j < 6; ++j) ss += sc[j] * sc[j];
    const float inv = 1.0f / fmaxf(sqrtf(ss), 1e-12f);

    _Float16* xr = xf16 + (size_t)row * K0P;
    #pragma unroll
    for (int i = 0; i < 20; ++i) xr[i * 16 + d] = (_Float16)(v[i] * inv);
    if (d < 6) xr[320 + d] = (_Float16)(sc[d] * inv);
    xr[326 + d] = (_Float16)0.0f;                 // zero pad cols 326..341
    if (d < 10) xr[342 + d] = (_Float16)0.0f;     // zero pad cols 342..351

    if (d == 0) {   // wide/linear logit (one lane per row)
        float ll = sc[0] + sc[1] + sc[2] + sc[3];
        int ix;
        ix = user_id[row]; if (ix) ll += l_user[ix];
        ix = shop_id[row]; if (ix) ll += l_shop[ix];
        ix = item_id[row]; if (ix) ll += l_item[ix];
        ix = cat1[row];    if (ix) ll += l_cat1[ix];
        ix = ttype[row];   if (ix) ll += l_tt[ix];
        const int* sl = shop_l + row * SEQ;
        #pragma unroll 5
        for (int i = 0; i < SEQ; ++i) { int j = sl[i]; if (j) ll += l_shopl[j]; }
        linlogit[row] = ll;
    }
}

// ---------------------------------------------------------------------------
// Stage 2: WMMA GEMM + bias + ReLU.   out[M,N](f16) = relu(A[M,K] @ W[N,K]^T + b)
// Block = 256 thr (8 waves) -> 128x64 output tile. W k-tile (64 rows x 32 k
// halfs = 4KB) is double-buffer staged in LDS via GLOBAL_LOAD_ASYNC_TO_LDS_B128
// (one b128 lane-chunk per thread), amortized across all 8 waves. Each wave
// owns a 16-row strip and 4 N-tiles: 1 A fragment feeds 4 WMMAs per k-step.
// A fragment (contiguous k, same order as B): lanes 0-15 row M=lane, K {0..15} of the
// k-step; lanes 16-31 same rows, K {16..31}.
// B fragment: lane n holds K 0..15 (lanes<16) / 16..31 -> one 32B LDS read.
// ---------------------------------------------------------------------------
__global__ void __launch_bounds__(256, 2) wdl_gemm_relu_wmma(
    const _Float16* __restrict__ A, const _Float16* __restrict__ W,
    const float* __restrict__ bias, _Float16* __restrict__ out,
    int M, int N, int K)
{
    __shared__ __align__(64) _Float16 lds[2][64][32];   // 8 KB double buffer

    const int tid  = threadIdx.x;
    const int lane = tid & 31;
    const int wave = tid >> 5;
    const int hlf  = lane >> 4;
    const int l15  = lane & 15;

    const int nbn = N >> 6;                         // 64-col tiles across N
    const int bmi = blockIdx.x / nbn;               // wave-uniform
    const int bm0 = bmi << 7;                       // 128-row tile origin
    const int bn0 = (blockIdx.x - bmi * nbn) << 6;  // 64-col tile origin

    // staging coords: 256 threads x 16B = 4KB = one k-tile of W
    const int srow = tid >> 2;                      // 0..63  (W row in tile)
    const int scol = (tid & 3) << 3;                // 0,8,16,24 (half offset)
    const _Float16* __restrict__ sgp = W + (size_t)(bn0 + srow) * K + scol;

    // compute coords
    const _Float16* __restrict__ arow =
        A + (size_t)(bm0 + (wave << 4) + l15) * K + (hlf << 4);

    const int nk = K >> 5;
    stage16(sgp, &lds[0][srow][scol]);              // preload k-tile 0

    v8f c0 = {}, c1 = {}, c2 = {}, c3 = {};
    for (int j = 0; j < nk; ++j) {
        const int cur = j & 1;
        wait_async0();                              // my staged chunk landed
        __syncthreads();                            // whole tile landed
        if (j + 1 < nk)                             // uniform branch
            stage16(sgp + (j + 1) * 32, &lds[1 - cur][srow][scol]);

        const int k0 = j << 5;
        __builtin_prefetch(arow + k0 + 32, 0, 1);   // global_prefetch_b8 (A stream)
        v8h alo = *(const v8h*)(arow + k0);
        v8h ahi = *(const v8h*)(arow + k0 + 8);
        v16h a;
        #pragma unroll
        for (int i = 0; i < 8; ++i) { a[i] = alo[i]; a[i + 8] = ahi[i]; }

        v16h b0 = *(const v16h*)(&lds[cur][ 0 + l15][hlf << 4]);
        v16h b1 = *(const v16h*)(&lds[cur][16 + l15][hlf << 4]);
        v16h b2 = *(const v16h*)(&lds[cur][32 + l15][hlf << 4]);
        v16h b3 = *(const v16h*)(&lds[cur][48 + l15][hlf << 4]);

        c0 = __builtin_amdgcn_wmma_f32_16x16x32_f16(false, a, false, b0, (short)0, c0, false, false);
        c1 = __builtin_amdgcn_wmma_f32_16x16x32_f16(false, a, false, b1, (short)0, c1, false, false);
        c2 = __builtin_amdgcn_wmma_f32_16x16x32_f16(false, a, false, b2, (short)0, c2, false, false);
        c3 = __builtin_amdgcn_wmma_f32_16x16x32_f16(false, a, false, b3, (short)0, c3, false, false);
        __syncthreads();                            // done reading buf 'cur'
    }

    // D layout: VGPR r -> M = r + 8*(lane>=16), N = lane&15
    const int orow = bm0 + (wave << 4) + (hlf << 3);
    v8f acc[4] = { c0, c1, c2, c3 };
    #pragma unroll
    for (int t = 0; t < 4; ++t) {
        const int ocol = bn0 + (t << 4) + l15;
        const float bv = bias[ocol];
        #pragma unroll
        for (int r = 0; r < 8; ++r) {
            float vo = fmaxf(acc[t][r] + bv, 0.0f);
            out[(size_t)(orow + r) * N + ocol] = (_Float16)vo;
        }
    }
}

// ---------------------------------------------------------------------------
// Stage 3: final 64-dot + bias + wide logit + sigmoid
// ---------------------------------------------------------------------------
__global__ void __launch_bounds__(256) wdl_final(
    const _Float16* __restrict__ h4, const float* __restrict__ W4,
    const float* __restrict__ b4, const float* __restrict__ linlogit,
    float* __restrict__ out, int M)
{
    int i = blockIdx.x * blockDim.x + threadIdx.x;
    if (i >= M) return;
    float acc = b4[0] + linlogit[i];
    const _Float16* h = h4 + (size_t)i * 64;
    #pragma unroll
    for (int k = 0; k < 64; ++k) acc += (float)h[k] * W4[k];
    out[i] = 1.0f / (1.0f + __expf(-acc));
}

// ---------------------------------------------------------------------------
extern "C" void kernel_launch(void* const* d_in, const int* in_sizes, int n_in,
                              void* d_out, int out_size, void* d_ws, size_t ws_size,
                              hipStream_t stream) {
    // ---- feature inputs (setup_inputs dict order) ----
    const int* user_id  = (const int*)d_in[0];
    const int* district = (const int*)d_in[1];
    const int* times_   = (const int*)d_in[2];
    const int* shop_id  = (const int*)d_in[3];
    const int* item_id  = (const int*)d_in[4];
    const int* cat1     = (const int*)d_in[5];
    const int* msf      = (const int*)d_in[6];
    const int* brand    = (const int*)d_in[7];
    const int* aoi      = (const int*)d_in[8];
    const int* geo6     = (const int*)d_in[9];
    const int* shop_l   = (const int*)d_in[10];
    const int* item_l   = (const int*)d_in[11];
    const int* cat1_l   = (const int*)d_in[12];
    const int* msf_l    = (const int*)d_in[13];
    const int* brand_l  = (const int*)d_in[14];
    const int* aoi_l    = (const int*)d_in[15];
    const int* geo6_l   = (const int*)d_in[16];
    const int* td_l     = (const int*)d_in[17];
    const int* ttype    = (const int*)d_in[18];
    const int* ttype_l  = (const int*)d_in[19];
    const float* rank7  = (const float*)d_in[20];
    const float* rank30 = (const float*)d_in[21];
    const float* rank90 = (const float*)d_in[22];
    const float* hours  = (const float*)d_in[23];
    const float* price_l= (const float*)d_in[24];
    const float* hours_l= (const float*)d_in[25];
    // ---- embedding tables ----
    const float* t_shop  = (const float*)d_in[26];
    const float* t_item  = (const float*)d_in[27];
    const float* t_cat1  = (const float*)d_in[28];
    const float* t_msf   = (const float*)d_in[29];
    const float* t_brand = (const float*)d_in[30];
    const float* t_aoi   = (const float*)d_in[31];
    const float* t_geo6  = (const float*)d_in[32];
    const float* t_user  = (const float*)d_in[33];
    const float* t_dist  = (const float*)d_in[34];
    const float* t_times = (const float*)d_in[35];
    const float* t_td    = (const float*)d_in[36];
    const float* t_tt    = (const float*)d_in[37];
    const float* t_ttl   = (const float*)d_in[38];
    // ---- DNN params ----
    const float* W0 = (const float*)d_in[39];   // [1024,326]
    const float* W1 = (const float*)d_in[40];   // [512,1024]
    const float* W2 = (const float*)d_in[41];   // [256,512]
    const float* W3 = (const float*)d_in[42];   // [64,256]
    const float* W4 = (const float*)d_in[43];   // [1,64]
    const float* b0 = (const float*)d_in[44];
    const float* b1 = (const float*)d_in[45];
    const float* b2 = (const float*)d_in[46];
    const float* b3 = (const float*)d_in[47];
    const float* b4 = (const float*)d_in[48];
    // ---- wide/linear tables ----
    const float* l_user  = (const float*)d_in[49];
    const float* l_shop  = (const float*)d_in[50];
    const float* l_item  = (const float*)d_in[51];
    const float* l_cat1  = (const float*)d_in[52];
    const float* l_shopl = (const float*)d_in[53];
    const float* l_tt    = (const float*)d_in[54];

    // ---- workspace layout ----
    char* ws = (char*)d_ws;
    _Float16* W0h = (_Float16*)ws;                       // 1024*352
    _Float16* W1h = W0h + 1024 * K0P;                    // 512*1024
    _Float16* W2h = W1h + 512 * 1024;                    // 256*512
    _Float16* W3h = W2h + 256 * 512;                     // 64*256
    float*    linlogit = (float*)(ws + (4ull  << 20));   // [B] f32
    _Float16* bufA     = (_Float16*)(ws + (8ull  << 20)); // 16 MB: x / h2 / h4
    _Float16* bufB     = (_Float16*)(ws + (24ull << 20)); // 32 MB: h1 / h3

    // Stage 0: weight conversion (+K padding for layer 0)
    auto cvt = [&](const float* s, _Float16* d, int N, int K, int Kp) {
        int tot = N * Kp;
        wdl_w2h<<<(tot + 255) / 256, 256, 0, stream>>>(s, d, N, K, Kp);
    };
    cvt(W0, W0h, 1024, 326, K0P);
    cvt(W1, W1h, 512, 1024, 1024);
    cvt(W2, W2h, 256, 512, 512);
    cvt(W3, W3h, 64, 256, 256);

    // Stage 1: gather + normalize + wide logit -> bufA (x_f16 [B,352])
    wdl_gather_normalize<<<BATCH / 8, 128, 0, stream>>>(
        user_id, district, times_, shop_id, item_id, cat1, msf, brand, aoi, geo6,
        shop_l, item_l, cat1_l, msf_l, brand_l, aoi_l, geo6_l, td_l,
        ttype, ttype_l,
        rank7, rank30, rank90, hours, price_l, hours_l,
        t_shop, t_item, t_cat1, t_msf, t_brand, t_aoi, t_geo6,
        t_user, t_dist, t_times, t_td, t_tt, t_ttl,
        l_user, l_shop, l_item, l_cat1, l_shopl, l_tt,
        bufA, linlogit);

    // Stage 2: WMMA MLP. Blocks = (M/128)*(N/64), 256 thr each.
    const int MB = BATCH / 128;   // 128 row blocks
    wdl_gemm_relu_wmma<<<MB * (1024/64), 256, 0, stream>>>(bufA, W0h, b0, bufB, BATCH, 1024, K0P);
    wdl_gemm_relu_wmma<<<MB * (512 /64), 256, 0, stream>>>(bufB, W1h, b1, bufA, BATCH, 512, 1024);
    wdl_gemm_relu_wmma<<<MB * (256 /64), 256, 0, stream>>>(bufA, W2h, b2, bufB, BATCH, 256, 512);
    wdl_gemm_relu_wmma<<<MB * (64  /64), 256, 0, stream>>>(bufB, W3h, b3, bufA, BATCH, 64, 256);

    // Stage 3: final dot + sigmoid
    wdl_final<<<BATCH / 256, 256, 0, stream>>>(bufA, W4, b4, linlogit, (float*)d_out, BATCH);
}
